// AttentionModel_16801912062366
// MI455X (gfx1250) — matrix operand order repaired and checked by reading, not compiled, not run on hardware
//
#include <hip/hip_runtime.h>
#include <hip/hip_bf16.h>
#include <math.h>
#include <stdint.h>

typedef __attribute__((ext_vector_type(16))) _Float16 v16h;
typedef __attribute__((ext_vector_type(8)))  float    v8f;
typedef __attribute__((ext_vector_type(4)))  float    v4f;
typedef __attribute__((ext_vector_type(4)))  unsigned int v4u;

#define NB  4
#define NT  1024
#define NE  512
#define NH  8
#define NHS 64
#define NL  4
#define NV  32000
#define NBT (NB*NT)

// LDS row pitch for staged A tiles: 512 + 8 halves = 260 dwords; 260 % 64 = 4, so the
// 16 rows a fragment read touches sit on banks 4r..4r+3 -> conflict-free ds_load_b128.
#define LPITCH 520

#define WMMA16(a, b, c) __builtin_amdgcn_wmma_f32_16x16x32_f16(false, a, false, b, (short)0, c, false, false)

// ---------------- WMMA fragment loaders (per CDNA5 ISA VGPR layouts) ----------------
// A (16x32 f16, MxK) from row-major A[M][K]: 2 contiguous 16B chunks/lane -> 2x b128.
__device__ __forceinline__ v16h load_a_frag(const _Float16* A, int lda, int row0, int k0) {
  const int lane = threadIdx.x & 31;
  const _Float16* p = A + (size_t)(row0 + (lane & 15)) * lda + (k0 + ((lane >> 4) << 3));
  v16h a;
#pragma unroll
  for (int e = 0; e < 8; ++e)  a[e] = p[e];
#pragma unroll
  for (int e = 8; e < 16; ++e) a[e] = p[e + 8];
  return a;
}

// Same A layout but sourced from a staged LDS tile with LPITCH row pitch.
__device__ __forceinline__ v16h load_a_frag_lds(const _Float16* As, int row0, int k0) {
  const int lane = threadIdx.x & 31;
  const _Float16* p = As + (row0 + (lane & 15)) * LPITCH + (k0 + ((lane >> 4) << 3));
  v16h a;
#pragma unroll
  for (int e = 0; e < 8; ++e)  a[e] = p[e];
#pragma unroll
  for (int e = 8; e < 16; ++e) a[e] = p[e + 8];
  return a;
}

// B (32x16 f16, KxN), element (k,n) = Bt[n*nstride + k] (k-contiguous / pre-transposed):
// 2 contiguous 16B chunks per lane (same k order as A) -> 2x b128.
__device__ __forceinline__ v16h load_bk1_frag(const _Float16* Bt, size_t nstride, int n0, int k0) {
  const int lane = threadIdx.x & 31;
  const _Float16* p = Bt + (size_t)(n0 + (lane & 15)) * nstride + (k0 + ((lane >> 4) << 3));
  v16h b;
#pragma unroll
  for (int e = 0; e < 8; ++e)  b[e] = p[e];
#pragma unroll
  for (int e = 8; e < 16; ++e) b[e] = p[e + 8];
  return b;
}

// Cooperative async copy of nrows x NE f16 rows into an LPITCH-padded LDS tile.
// Uses the CDNA5 async global->LDS engine: one instruction per 16B chunk, no VGPR
// data movement, drained with s_wait_asynccnt before the staging barrier.
__device__ __forceinline__ void stage_a_lds(const _Float16* __restrict__ src, int ld, int row0,
                                            _Float16* __restrict__ dst, int nrows) {
  const int nchunk = nrows * (NE / 8);     // 16B chunks
  for (int c = threadIdx.x; c < nchunk; c += blockDim.x) {
    const int row = c >> 6, col = (c & 63) << 3;
    // LDS aperture keeps the byte offset in ADDR[31:0]; hardware adds LDS_BASE.
    const unsigned lds_off = (unsigned)(uintptr_t)(dst + row * LPITCH + col);
    const _Float16* g = src + (size_t)(row0 + row) * ld + col;
    asm volatile("global_load_async_to_lds_b128 %0, %1, off"
                 :: "v"(lds_off), "v"(g) : "memory");
  }
  asm volatile("s_wait_asynccnt 0x0" ::: "memory");
}

// ---------------- LDS-A, double-buffered-B 32x64-per-wave GEMM core ----------------
// Per k-step: 4 global B-frag loads for k+32 stay in flight under 8 WMMAs; A comes
// from the block-shared LDS tile (cheap, single-buffered).
__device__ __forceinline__ void gemm_lds_2x4(const _Float16* As,
                                             const _Float16* __restrict__ Bt, int ldb, int n0,
                                             v8f c[2][4]) {
  v16h b0 = load_bk1_frag(Bt, ldb, n0,      0);
  v16h b1 = load_bk1_frag(Bt, ldb, n0 + 16, 0);
  v16h b2 = load_bk1_frag(Bt, ldb, n0 + 32, 0);
  v16h b3 = load_bk1_frag(Bt, ldb, n0 + 48, 0);
  for (int k0 = 0; k0 < NE; k0 += 32) {
    const int kn = (k0 + 32 < NE) ? (k0 + 32) : 0;   // last-iter prefetch wraps (harmless)
    v16h b0n = load_bk1_frag(Bt, ldb, n0,      kn);
    v16h b1n = load_bk1_frag(Bt, ldb, n0 + 16, kn);
    v16h b2n = load_bk1_frag(Bt, ldb, n0 + 32, kn);
    v16h b3n = load_bk1_frag(Bt, ldb, n0 + 48, kn);
    const v16h a0 = load_a_frag_lds(As, 0,  k0);
    const v16h a1 = load_a_frag_lds(As, 16, k0);
    c[0][0] = WMMA16(a0, b0, c[0][0]);
    c[0][1] = WMMA16(a0, b1, c[0][1]);
    c[0][2] = WMMA16(a0, b2, c[0][2]);
    c[0][3] = WMMA16(a0, b3, c[0][3]);
    c[1][0] = WMMA16(a1, b0, c[1][0]);
    c[1][1] = WMMA16(a1, b1, c[1][1]);
    c[1][2] = WMMA16(a1, b2, c[1][2]);
    c[1][3] = WMMA16(a1, b3, c[1][3]);
    b0 = b0n; b1 = b1n; b2 = b2n; b3 = b3n;
  }
}

// ---------------- fused cast fp32 -> f16 + transpose: in[K,N] -> out[N,K] (batched) ----
__global__ void cvt_transpose_kernel(const float* __restrict__ in, _Float16* __restrict__ out,
                                     int K, int N) {
  __shared__ _Float16 tile[32][33];
  const size_t base = (size_t)blockIdx.z * K * N;
  const int n0 = blockIdx.x * 32, k0 = blockIdx.y * 32;
  const int tx = threadIdx.x & 31, ty = threadIdx.x >> 5;   // 32 x 8
#pragma unroll
  for (int r = ty; r < 32; r += 8)
    tile[r][tx] = (_Float16)in[base + (size_t)(k0 + r) * N + (n0 + tx)];
  __syncthreads();
#pragma unroll
  for (int r = ty; r < 32; r += 8)
    out[base + (size_t)(n0 + r) * K + (k0 + tx)] = tile[tx][r];
}

// ---------------- embedding: x = tok_emb[idx] + pos_emb ----------------
__global__ void embed_kernel(const int* __restrict__ idx, const float* __restrict__ tok,
                             const float* __restrict__ pos, float* __restrict__ x) {
  const int row = blockIdx.x;             // 0..NBT-1
  const int t = row & (NT - 1);
  const int token = idx[row];
  for (int e = threadIdx.x; e < NE; e += blockDim.x)
    x[(size_t)row * NE + e] = tok[(size_t)token * NE + e] + pos[(size_t)t * NE + e];
}

// ---------------- layernorm (fp32 in, f16 out) ----------------
__global__ void layernorm_kernel(const float* __restrict__ x, const float* __restrict__ g,
                                 const float* __restrict__ bta, _Float16* __restrict__ out) {
  const int row = blockIdx.x;
  const int tid = threadIdx.x;            // 256 threads, 2 elems each (E=512)
  const float* xr = x + (size_t)row * NE;
  const float v0 = xr[tid], v1 = xr[tid + 256];
  float s1 = v0 + v1, s2 = v0 * v0 + v1 * v1;
#pragma unroll
  for (int off = 16; off; off >>= 1) {
    s1 += __shfl_xor(s1, off, 32);
    s2 += __shfl_xor(s2, off, 32);
  }
  __shared__ float sh1[8], sh2[8];
  if ((tid & 31) == 0) { sh1[tid >> 5] = s1; sh2[tid >> 5] = s2; }
  __syncthreads();
  float t1 = 0.f, t2 = 0.f;
#pragma unroll
  for (int w = 0; w < 8; ++w) { t1 += sh1[w]; t2 += sh2[w]; }
  const float mu   = t1 * (1.0f / NE);
  const float var  = t2 * (1.0f / NE) - mu * mu;
  const float rstd = rsqrtf(var + 1e-5f);
  out[(size_t)row * NE + tid]       = (_Float16)((v0 - mu) * rstd * g[tid]       + bta[tid]);
  out[(size_t)row * NE + tid + 256] = (_Float16)((v1 - mu) * rstd * g[tid + 256] + bta[tid + 256]);
}

// ---------------- q/k/v projection: 16x64 slab per wave, fully double-buffered -------
// storeT==0 -> out[B,H,T,HS] (q,k); storeT==1 -> out[B,H,HS,T] (V transposed for P*V).
__global__ void qkv_gemm_kernel(const _Float16* __restrict__ h, const _Float16* __restrict__ wt,
                                _Float16* __restrict__ out, float scale, int storeT) {
  const int wave = threadIdx.x >> 5;
  const int head = blockIdx.y;
  const int mt = blockIdx.x * 8 + wave;   // 0..255
  const int row0 = mt * 16;
  const _Float16* Wt = wt + (size_t)head * NHS * NE;   // [HS,E]
  v8f c0 = {}, c1 = {}, c2 = {}, c3 = {};
  v16h a  = load_a_frag(h, NE, row0, 0);
  v16h b0 = load_bk1_frag(Wt, NE, 0,  0);
  v16h b1 = load_bk1_frag(Wt, NE, 16, 0);
  v16h b2 = load_bk1_frag(Wt, NE, 32, 0);
  v16h b3 = load_bk1_frag(Wt, NE, 48, 0);
  for (int k0 = 0; k0 < NE; k0 += 32) {
    const int kn = (k0 + 32 < NE) ? (k0 + 32) : 0;
    v16h an  = load_a_frag(h, NE, row0, kn);
    v16h b0n = load_bk1_frag(Wt, NE, 0,  kn);
    v16h b1n = load_bk1_frag(Wt, NE, 16, kn);
    v16h b2n = load_bk1_frag(Wt, NE, 32, kn);
    v16h b3n = load_bk1_frag(Wt, NE, 48, kn);
    c0 = WMMA16(a, b0, c0);
    c1 = WMMA16(a, b1, c1);
    c2 = WMMA16(a, b2, c2);
    c3 = WMMA16(a, b3, c3);
    a = an; b0 = b0n; b1 = b1n; b2 = b2n; b3 = b3n;
  }
  const int lane = threadIdx.x & 31;
  const int colh = lane & 15;
  const int rb   = row0 + ((lane >> 4) << 3);
  if (!storeT) {
#pragma unroll
    for (int r = 0; r < 8; ++r) {
      const int m = rb + r;
      const int bb = m >> 10, t = m & (NT - 1);
      _Float16* o = out + (((size_t)bb * NH + head) * NT + t) * NHS + colh;
      o[0]  = (_Float16)(c0[r] * scale);
      o[16] = (_Float16)(c1[r] * scale);
      o[32] = (_Float16)(c2[r] * scale);
      o[48] = (_Float16)(c3[r] * scale);
    }
  } else {
#pragma unroll
    for (int r = 0; r < 8; ++r) {
      const int m = rb + r;
      const int bb = m >> 10, t = m & (NT - 1);
      _Float16* o = out + ((size_t)bb * NH + head) * NHS * NT + t;
      o[(size_t)(colh)      * NT] = (_Float16)(c0[r] * scale);
      o[(size_t)(colh + 16) * NT] = (_Float16)(c1[r] * scale);
      o[(size_t)(colh + 32) * NT] = (_Float16)(c2[r] * scale);
      o[(size_t)(colh + 48) * NT] = (_Float16)(c3[r] * scale);
    }
  }
}

// ---------------- flash attention: one wave per 16-row Q tile ----------------
__global__ void attn_kernel(const _Float16* __restrict__ q, const _Float16* __restrict__ k,
                            const _Float16* __restrict__ vT, _Float16* __restrict__ out) {
  const int qt = blockIdx.x;               // 0..63
  const int bh = blockIdx.y;               // 0..31
  const int bb = bh >> 3, hh = bh & 7;
  const _Float16* Q  = q  + (size_t)bh * NT * NHS;
  const _Float16* K  = k  + (size_t)bh * NT * NHS;
  const _Float16* Vt = vT + (size_t)bh * NHS * NT;
  const int lane  = threadIdx.x & 31;
  const int row0  = qt * 16;
  const int rbase = (lane >> 4) << 3;
  const int colh  = lane & 15;

  __shared__ _Float16 pld[16][32];         // P tile staged for C-layout -> A-layout

  const v16h qa0 = load_a_frag(Q, NHS, row0, 0);   // Q pre-scaled by E^-0.5
  const v16h qa1 = load_a_frag(Q, NHS, row0, 32);

  v16h kf0 = load_bk1_frag(K, NHS, 0,  0);
  v16h kf1 = load_bk1_frag(K, NHS, 0,  32);
  v16h kf2 = load_bk1_frag(K, NHS, 16, 0);
  v16h kf3 = load_bk1_frag(K, NHS, 16, 32);

  v8f o0 = {}, o1 = {}, o2 = {}, o3 = {};
  float m[8], l[8];
#pragma unroll
  for (int r = 0; r < 8; ++r) { m[r] = -INFINITY; l[r] = 0.f; }

  for (int kb = 0; kb < row0 + 16; kb += 32) {     // 32 keys per block
    v8f s0 = {}, s1 = {};
    s0 = WMMA16(qa0, kf0, s0);
    s0 = WMMA16(qa1, kf1, s0);
    s1 = WMMA16(qa0, kf2, s1);
    s1 = WMMA16(qa1, kf3, s1);

    // issue V frags (this block) and K frags (next block) before softmax
    const v16h vf0 = load_bk1_frag(Vt, NT, 0,  kb);
    const v16h vf1 = load_bk1_frag(Vt, NT, 16, kb);
    const v16h vf2 = load_bk1_frag(Vt, NT, 32, kb);
    const v16h vf3 = load_bk1_frag(Vt, NT, 48, kb);
    const int kbn = (kb + 32 < row0 + 16) ? (kb + 32) : 0;
    kf0 = load_bk1_frag(K, NHS, kbn,      0);
    kf1 = load_bk1_frag(K, NHS, kbn,      32);
    kf2 = load_bk1_frag(K, NHS, kbn + 16, 0);
    kf3 = load_bk1_frag(K, NHS, kbn + 16, 32);

    if (kb + 31 > row0) {                  // block touches the causal diagonal
      const int gc0 = kb + colh, gc1 = gc0 + 16;
#pragma unroll
      for (int r = 0; r < 8; ++r) {
        const int gr = row0 + rbase + r;
        if (gc0 > gr) s0[r] = -INFINITY;
        if (gc1 > gr) s1[r] = -INFINITY;
      }
    }
#pragma unroll
    for (int r = 0; r < 8; ++r) {
      float mx = fmaxf(s0[r], s1[r]);
#pragma unroll
      for (int off = 8; off; off >>= 1) mx = fmaxf(mx, __shfl_xor(mx, off, 32));
      const float mn   = fmaxf(m[r], mx);
      const float corr = __expf(m[r] - mn);
      m[r] = mn;
      const float p0 = __expf(s0[r] - mn);
      const float p1 = __expf(s1[r] - mn);
      float ps = p0 + p1;
#pragma unroll
      for (int off = 8; off; off >>= 1) ps += __shfl_xor(ps, off, 32);
      l[r] = l[r] * corr + ps;
      o0[r] *= corr; o1[r] *= corr; o2[r] *= corr; o3[r] *= corr;
      pld[rbase + r][colh]      = (_Float16)p0;
      pld[rbase + r][colh + 16] = (_Float16)p1;
    }
    __syncthreads();                        // single-wave WG: wave-local LDS fence
    const v16h pa = load_a_frag(&pld[0][0], 32, 0, 0);
    __syncthreads();
    o0 = WMMA16(pa, vf0, o0);
    o1 = WMMA16(pa, vf1, o1);
    o2 = WMMA16(pa, vf2, o2);
    o3 = WMMA16(pa, vf3, o3);
  }
  // write out[b, t, h*HS + d]  (== transpose(0,2,1,3).reshape)
#pragma unroll
  for (int r = 0; r < 8; ++r) {
    const float invl = 1.0f / l[r];
    _Float16* orow = out + ((size_t)bb * NT + row0 + rbase + r) * NE + hh * NHS + colh;
    orow[0]  = (_Float16)(o0[r] * invl);
    orow[16] = (_Float16)(o1[r] * invl);
    orow[32] = (_Float16)(o2[r] * invl);
    orow[48] = (_Float16)(o3[r] * invl);
  }
}

// ---------------- out-proj + bias + residual: x += attn @ Wproj + b ----------------
// Block = 8 waves sharing one 32-row A tile in LDS; wave w covers n-group w.
__global__ void proj_gemm_kernel(const _Float16* __restrict__ a, const _Float16* __restrict__ wt,
                                 const float* __restrict__ bias, float* __restrict__ x) {
  __shared__ _Float16 As[32 * LPITCH];
  const int mg = blockIdx.x;                // 0..127
  const int row0 = mg * 32;
  stage_a_lds(a, NE, row0, As, 32);
  __syncthreads();
  const int n0 = (threadIdx.x >> 5) * 64;   // wave -> n-group
  v8f c[2][4] = {};
  gemm_lds_2x4(As, wt, NE, n0, c);
  const int lane = threadIdx.x & 31;
  const int col  = n0 + (lane & 15);
  const int rbo  = (lane >> 4) << 3;
#pragma unroll
  for (int mi = 0; mi < 2; ++mi) {
    const int rb = row0 + mi * 16 + rbo;
#pragma unroll
    for (int r = 0; r < 8; ++r) {
      float* xr = x + (size_t)(rb + r) * NE + col;
      xr[0]  += c[mi][0][r] + bias[col];
      xr[16] += c[mi][1][r] + bias[col + 16];
      xr[32] += c[mi][2][r] + bias[col + 32];
      xr[48] += c[mi][3][r] + bias[col + 48];
    }
  }
}

// ---------------- LM head: logits = h @ Wlm + blm (fp32, NT stores) ----------------
// Block = 4 waves sharing one 32-row A tile in LDS; grid (125,128) -> exactly 500
// n-groups x 128 m-groups, no guards, uniform barrier.
__global__ void lmhead_gemm_kernel(const _Float16* __restrict__ a, const _Float16* __restrict__ wt,
                                   const float* __restrict__ bias, float* __restrict__ logits) {
  __shared__ _Float16 As[32 * LPITCH];
  const int mg = blockIdx.y;                // 0..127 (32 rows each)
  const int row0 = mg * 32;
  stage_a_lds(a, NE, row0, As, 32);
  __syncthreads();
  const int ng = blockIdx.x * 4 + (threadIdx.x >> 5);   // 0..499
  const int n0 = ng * 64;
  v8f c[2][4] = {};
  gemm_lds_2x4(As, wt, NE, n0, c);
  const int lane = threadIdx.x & 31;
  const int col  = n0 + (lane & 15);
  const int rbo  = (lane >> 4) << 3;
#pragma unroll
  for (int mi = 0; mi < 2; ++mi) {
    const int rb = row0 + mi * 16 + rbo;
#pragma unroll
    for (int r = 0; r < 8; ++r) {
      float* lr = logits + (size_t)(rb + r) * NV + col;
      __builtin_nontemporal_store(c[mi][0][r] + bias[col],      lr);
      __builtin_nontemporal_store(c[mi][1][r] + bias[col + 16], lr + 16);
      __builtin_nontemporal_store(c[mi][2][r] + bias[col + 32], lr + 32);
      __builtin_nontemporal_store(c[mi][3][r] + bias[col + 48], lr + 48);
    }
  }
}

// ---------------- per-row NLL: single-pass online logsumexp - logit[target] ----------
__global__ void nll_kernel(const float* __restrict__ logits, const int* __restrict__ tgt,
                           float* __restrict__ nll) {
  const int row = blockIdx.x;
  const int tid = threadIdx.x;              // 256
  const float* lr = logits + (size_t)row * NV;
  float m = -INFINITY, se = 0.f;
  for (int i = tid * 4; i < NV; i += 1024) {
    const v4f v = __builtin_nontemporal_load((const v4f*)(lr + i));
    const float bm = fmaxf(fmaxf(v.x, v.y), fmaxf(v.z, v.w));
    const float mn = fmaxf(m, bm);
    se = se * __expf(m - mn)
       + __expf(v.x - mn) + __expf(v.y - mn) + __expf(v.z - mn) + __expf(v.w - mn);
    m = mn;
  }
#pragma unroll
  for (int off = 16; off; off >>= 1) {
    const float mo = __shfl_xor(m, off, 32);
    const float so = __shfl_xor(se, off, 32);
    const float mn = fmaxf(m, mo);
    se = se * __expf(m - mn) + so * __expf(mo - mn);
    m = mn;
  }
  __shared__ float shm[8], shs[8];
  if ((tid & 31) == 0) { shm[tid >> 5] = m; shs[tid >> 5] = se; }
  __syncthreads();
  if (tid == 0) {
    float M = shm[0], S = shs[0];
#pragma unroll
    for (int w = 1; w < 8; ++w) {
      const float mn = fmaxf(M, shm[w]);
      S = S * __expf(M - mn) + shs[w] * __expf(shm[w] - mn);
      M = mn;
    }
    nll[row] = (M + __logf(S)) - lr[tgt[row]];
  }
}

__global__ void loss_reduce_kernel(const float* __restrict__ nll, float* __restrict__ loss) {
  const int tid = threadIdx.x;              // 256
  float s = 0.f;
  for (int i = tid; i < NBT; i += 256) s += nll[i];
#pragma unroll
  for (int off = 16; off; off >>= 1) s += __shfl_xor(s, off, 32);
  __shared__ float sh[8];
  if ((tid & 31) == 0) sh[tid >> 5] = s;
  __syncthreads();
  if (tid == 0) {
    float t = 0.f;
#pragma unroll
    for (int w = 0; w < 8; ++w) t += sh[w];
    loss[0] = t * (1.0f / NBT);
  }
}

// ---------------- host orchestration ----------------
extern "C" void kernel_launch(void* const* d_in, const int* in_sizes, int n_in,
                              void* d_out, int out_size, void* d_ws, size_t ws_size,
                              hipStream_t stream) {
  (void)in_sizes; (void)n_in; (void)out_size; (void)ws_size;
  const int*   idx   = (const int*)  d_in[0];
  const int*   tgt   = (const int*)  d_in[1];
  const float* tok   = (const float*)d_in[2];
  const float* pos   = (const float*)d_in[3];
  const float* Wq    = (const float*)d_in[4];
  const float* Wk    = (const float*)d_in[5];
  const float* Wv    = (const float*)d_in[6];
  const float* Wproj = (const float*)d_in[7];
  const float* bproj = (const float*)d_in[8];
  const float* ln1g  = (const float*)d_in[9];
  const float* ln1b  = (const float*)d_in[10];
  const float* lnfg  = (const float*)d_in[11];
  const float* lnfb  = (const float*)d_in[12];
  const float* Wlm   = (const float*)d_in[13];
  const float* blm   = (const float*)d_in[14];

  float* logits = (float*)d_out;
  float* loss   = logits + (size_t)NBT * NV;

  // workspace carve-up (~71 MB); all weight copies are TRANSPOSED (N-major, k-contiguous)
  char* w = (char*)d_ws;
  const size_t nWqkv  = (size_t)NL * NH * NE * NHS;   // 1,048,576
  const size_t nWproj = (size_t)NL * NE * NE;         // 1,048,576
  const size_t nWlm   = (size_t)NE * NV;              // 16,384,000
  _Float16* wq16 = (_Float16*)w;  w += nWqkv  * 2;    // [L,H,HS,E]
  _Float16* wk16 = (_Float16*)w;  w += nWqkv  * 2;    // [L,H,HS,E]
  _Float16* wv16 = (_Float16*)w;  w += nWqkv  * 2;    // [L,H,HS,E]
  _Float16* wp16 = (_Float16*)w;  w += nWproj * 2;    // [L,E,E] transposed
  _Float16* wl16 = (_Float16*)w;  w += nWlm   * 2;    // [V,E]
  float*    x    = (float*)w;     w += (size_t)NBT * NE * 4;
  _Float16* h16  = (_Float16*)w;  w += (size_t)NBT * NE * 2;
  _Float16* q16  = (_Float16*)w;  w += (size_t)NBT * NE * 2;   // [B,H,T,HS]
  _Float16* k16  = (_Float16*)w;  w += (size_t)NBT * NE * 2;   // [B,H,T,HS]
  _Float16* vT16 = (_Float16*)w;  w += (size_t)NBT * NE * 2;   // [B,H,HS,T]
  _Float16* at16 = (_Float16*)w;  w += (size_t)NBT * NE * 2;
  float*    nllb = (float*)w;     w += (size_t)NBT * 4;

  // cast + transpose all weights once (cheap vs the 524MB logits stream)
  cvt_transpose_kernel<<<dim3(NHS/32, NE/32, NL*NH), 256, 0, stream>>>(Wq,    wq16, NE, NHS);
  cvt_transpose_kernel<<<dim3(NHS/32, NE/32, NL*NH), 256, 0, stream>>>(Wk,    wk16, NE, NHS);
  cvt_transpose_kernel<<<dim3(NHS/32, NE/32, NL*NH), 256, 0, stream>>>(Wv,    wv16, NE, NHS);
  cvt_transpose_kernel<<<dim3(NE/32,  NE/32, NL),    256, 0, stream>>>(Wproj, wp16, NE, NE);
  cvt_transpose_kernel<<<dim3(NV/32,  NE/32, 1),     256, 0, stream>>>(Wlm,   wl16, NE, NV);

  embed_kernel<<<NBT, 256, 0, stream>>>(idx, tok, pos, x);

  const float qscale = 0.044194173824159216f;  // E^-0.5 = 512^-0.5
  for (int l = 0; l < NL; ++l) {
    layernorm_kernel<<<NBT, 256, 0, stream>>>(x, ln1g + (size_t)l * NE, ln1b + (size_t)l * NE, h16);
    const _Float16* wql = wq16 + (size_t)l * NH * NHS * NE;
    const _Float16* wkl = wk16 + (size_t)l * NH * NHS * NE;
    const _Float16* wvl = wv16 + (size_t)l * NH * NHS * NE;
    dim3 gq(32, NH);
    qkv_gemm_kernel<<<gq, 256, 0, stream>>>(h16, wql, q16,  qscale, 0);
    qkv_gemm_kernel<<<gq, 256, 0, stream>>>(h16, wkl, k16,  1.0f,   0);
    qkv_gemm_kernel<<<gq, 256, 0, stream>>>(h16, wvl, vT16, 1.0f,   1);
    dim3 ga(NT / 16, NB * NH);
    attn_kernel<<<ga, 32, 0, stream>>>(q16, k16, vT16, at16);
    proj_gemm_kernel<<<128, 256, 0, stream>>>(at16, wp16 + (size_t)l * NE * NE,
                                              bproj + (size_t)l * NE, x);
  }
  layernorm_kernel<<<NBT, 256, 0, stream>>>(x, lnfg, lnfb, h16);
  dim3 gl(125, 128);
  lmhead_gemm_kernel<<<gl, 128, 0, stream>>>(h16, wl16, blm, logits);
  nll_kernel<<<NBT, 256, 0, stream>>>(logits, tgt, nllb);
  loss_reduce_kernel<<<1, 256, 0, stream>>>(nllb, loss);
}
